// HandMadeTrainableRNN_91036126806942
// MI455X (gfx1250) — matrix operand order repaired and checked by reading, not compiled, not run on hardware
//
#include <hip/hip_runtime.h>
#include <hip/hip_bf16.h>
#include <math.h>

// ---------------------------------------------------------------------------
// HandMadeTrainableRNN on MI455X (gfx1250, wave32, WMMA)
//
// Per step: rW = r @ w_rec   ([64,2048] x [2048,2048], bf16 WMMA, f32 acc)
//           h  = 0.8h + 0.2*rW + 0.05*noise_t + 0.2*x_t*wi ; r = tanh(h)
//           out[b,t] = sum_j r[b,j]*wo[j] / 2048
//
// 32 persistent workgroups, each owns 64 columns of H=2048:
//   - w_rec column slice staged once in LDS as bf16 (64 x 2048, padded stride)
//   - h lives in WMMA accumulator registers for all 512 steps
//   - r ping-pongs between two L2-resident bf16 buffers in workspace
//   - K reduction: 4 independent WMMA accumulators; A operand double-buffered
//     in groups of 4 chunks with sched_barrier(0) fences so the next group's
//     global loads are issued before the current group's WMMAs (hides L2
//     latency instead of wait-to-zero before every WMMA)
//   - device-wide barrier per step via atomic counter + s_sleep
// ---------------------------------------------------------------------------

typedef __attribute__((ext_vector_type(16))) __bf16 v16bf;
typedef __attribute__((ext_vector_type(8)))  __bf16 v8bf;
typedef __attribute__((ext_vector_type(8)))  float  v8f;

#define Bsz 64
#define Ssz 512
#define Hsz 2048
#define NWG 32
#define COLS_PER_WG 64
#define LDSW_STRIDE 2056      // 2048 + 8 bf16 pad -> 4112B col stride, kills 16-way bank conflicts
#define LDS_BYTES (COLS_PER_WG * LDSW_STRIDE * 2)

__global__ __launch_bounds__(256) void rnn_setup(
    const float* __restrict__ wi, const float* __restrict__ wo,
    const float* __restrict__ A,
    const float* __restrict__ ovwi, const float* __restrict__ ovwo,
    const float* __restrict__ ovmn, const float* __restrict__ h0,
    float* __restrict__ n_rec, float* __restrict__ m_rec,
    __bf16* __restrict__ rbuf0, float* __restrict__ out,
    unsigned* __restrict__ gcount)
{
  __shared__ float s1[256], s2[256];
  const int tid = threadIdx.x;
  float swi = 0.f, swo = 0.f;
  for (int k = tid; k < Hsz; k += 256) {
    float a = wi[k], b = wo[k];
    swi += a * a; swo += b * b;
  }
  s1[tid] = swi; s2[tid] = swo;
  __syncthreads();
  for (int off = 128; off > 0; off >>= 1) {
    if (tid < off) { s1[tid] += s1[tid + off]; s2[tid] += s2[tid + off]; }
    __syncthreads();
  }
  const float Hf = (float)Hsz;
  const float ci = ovwi[0] * Hf / s1[0];
  const float co = ovwo[0] * Hf / s2[0];
  const float cm = sqrtf(ovmn[0] * Hf);
  for (int k = tid; k < Hsz; k += 256) {
    n_rec[k] = ci * wi[k] + cm * A[k];
    m_rec[k] = co * wo[k] + cm * A[k];
  }
  // r(t=-1) = tanh(h0) broadcast over batch
  for (int idx = tid; idx < Bsz * Hsz; idx += 256) {
    int j = idx & (Hsz - 1);
    rbuf0[idx] = (__bf16)tanhf(h0[j]);
  }
  // d_out is poisoned by harness; we accumulate with atomics -> zero it here.
  for (int idx = tid; idx < Bsz * Ssz; idx += 256) out[idx] = 0.f;
  if (tid == 0) *gcount = 0u;
}

__global__ __launch_bounds__(512, 1) void rnn_persistent(
    const float* __restrict__ x, const float* __restrict__ noise,
    const float* __restrict__ wi, const float* __restrict__ wo,
    const float* __restrict__ rec_noise, const float* __restrict__ h0,
    const float* __restrict__ n_rec, const float* __restrict__ m_rec,
    __bf16* __restrict__ rbuf, unsigned* __restrict__ gcount,
    float* __restrict__ out)
{
  extern __shared__ __bf16 lds_w[];        // [COLS_PER_WG][LDSW_STRIDE], column-major (K contiguous)
  __shared__ float lds_out[Bsz];

  const int tid  = threadIdx.x;
  const int jb   = blockIdx.x * COLS_PER_WG;       // this WG's global column base
  const float invH = 1.0f / (float)Hsz;

  // ---- stage w_rec slice into LDS once: w = rec_noise + n_rec*m_rec/H (bf16)
  {
    const int jl = tid & 63;
    const float mj = m_rec[jb + jl] * invH;
    for (int k = tid >> 6; k < Hsz; k += 8) {
      float w = rec_noise[(size_t)k * Hsz + (jb + jl)] + n_rec[k] * mj;
      lds_w[jl * LDSW_STRIDE + k] = (__bf16)w;
    }
  }
  if (tid < Bsz) lds_out[tid] = 0.f;

  const int lane  = tid & 31;
  const int wave  = tid >> 5;       // 16 waves: 4 mtiles x 4 ntiles of 16x16
  const int mtile = wave & 3;
  const int ntile = wave >> 2;
  const int half  = lane >> 4;      // lane half (0/1)
  const int l15   = lane & 15;

  const int jl    = ntile * 16 + l15;       // local col (B operand N / C,D N)
  const int j     = jb + jl;                // global col
  const int mbase = mtile * 16 + half * 8;  // C/D row base for this lane

  const float wij = wi[j];
  const float woj = wo[j];

  v8f h;                                    // h[b, j] held in accumulator layout
  { const float h0j = h0[j];
    for (int i = 0; i < 8; ++i) h[i] = h0j; }

  __syncthreads();

  // A (16x32 bf16): lane row = arow; lanes<16 K 0-7 & 16-23, lanes>=16 K 8-15 & 24-31
  const int arow  = mtile * 16 + l15;
  const int koffA = half * 8;
  // B (32x16 bf16): col N=lane; lanes<16 K 0-7 & 16-23, lanes>=16 K 8-15 & 24-31
  const __bf16* bBase = lds_w + jl * LDSW_STRIDE + half * 8;

  for (int t = 0; t < Ssz; ++t) {
    const __bf16* rd = rbuf + (size_t)(t & 1) * (Bsz * Hsz);
    __bf16*       wr = rbuf + (size_t)((t + 1) & 1) * (Bsz * Hsz);

    // ---- GEMM: c = r(t-1) @ Wslice, K = 2048 in 64 chunks of 32.
    // Chunks grouped by 4; A double-buffered one group ahead, fenced with
    // sched_barrier so the scheduler cannot sink next-group loads past the
    // current group's WMMAs. 4 independent accumulators (one per chunk slot).
    v8f c0 = {}, c1 = {}, c2 = {}, c3 = {};
    const __bf16* aBase = rd + (size_t)arow * Hsz + koffA;

    v8bf pA0[4], pA1[4], qA0[4], qA1[4];   // double-buffered A staging

#define LOADA(B0, B1, G)                                                       \
    {                                                                          \
      _Pragma("unroll")                                                        \
      for (int s = 0; s < 4; ++s) {                                            \
        B0[s] = *(const v8bf*)(aBase + ((G) * 4 + s) * 32);                    \
        B1[s] = *(const v8bf*)(aBase + ((G) * 4 + s) * 32 + 16);               \
      }                                                                        \
    }
#define WMMAC(ACC, A0, A1, KC)                                                 \
    {                                                                          \
      v8bf b0 = *(const v8bf*)(bBase + (KC) * 32);                             \
      v8bf b1 = *(const v8bf*)(bBase + (KC) * 32 + 16);                        \
      v16bf a = __builtin_shufflevector(A0, A1,                                \
                  0,1,2,3,4,5,6,7,8,9,10,11,12,13,14,15);                      \
      v16bf b = __builtin_shufflevector(b0, b1,                                \
                  0,1,2,3,4,5,6,7,8,9,10,11,12,13,14,15);                      \
      ACC = __builtin_amdgcn_wmma_f32_16x16x32_bf16(false, a, false, b,        \
                                                    (short)0, ACC, false, false); \
    }
#define CONSUME(B0, B1, G)                                                     \
    {                                                                          \
      WMMAC(c0, B0[0], B1[0], (G) * 4 + 0)                                     \
      WMMAC(c1, B0[1], B1[1], (G) * 4 + 1)                                     \
      WMMAC(c2, B0[2], B1[2], (G) * 4 + 2)                                     \
      WMMAC(c3, B0[3], B1[3], (G) * 4 + 3)                                     \
    }

    LOADA(pA0, pA1, 0)
#pragma unroll
    for (int g = 0; g < 16; g += 2) {
      LOADA(qA0, qA1, g + 1)                     // next group's A in flight
      __builtin_amdgcn_sched_barrier(0);
      CONSUME(pA0, pA1, g)
      __builtin_amdgcn_sched_barrier(0);
      if (g + 2 < 16) LOADA(pA0, pA1, g + 2)
      __builtin_amdgcn_sched_barrier(0);
      CONSUME(qA0, qA1, g + 1)
      __builtin_amdgcn_sched_barrier(0);
    }
#undef LOADA
#undef WMMAC
#undef CONSUME

    v8f c;
    for (int i = 0; i < 8; ++i) c[i] = (c0[i] + c1[i]) + (c2[i] + c3[i]);

    // ---- elementwise recurrence + readout partials
    for (int i = 0; i < 8; ++i) {
      const int b = mbase + i;
      const float nz = noise[((size_t)b * Ssz + t) * Hsz + j];
      const float xv = x[b * Ssz + t];
      float hn = 0.8f * h[i] + 0.2f * c[i] + 0.05f * nz + 0.2f * xv * wij;
      h[i] = hn;
      const float e  = __expf(2.f * hn);            // tanh via exp
      const float rv = 1.f - 2.f / (e + 1.f);
      wr[(size_t)b * Hsz + j] = (__bf16)rv;

      float v = rv * woj;                            // reduce over 16 cols (half-wave)
      v += __shfl_xor(v, 1, 32);
      v += __shfl_xor(v, 2, 32);
      v += __shfl_xor(v, 4, 32);
      v += __shfl_xor(v, 8, 32);
      if (l15 == 0) atomicAdd(&lds_out[b], v);       // ds_add_f32

      if (t + 1 < Ssz)                               // global_prefetch_b8 next noise slice
        __builtin_prefetch(&noise[((size_t)b * Ssz + (t + 1)) * Hsz + j], 0, 1);
    }

    __threadfence();          // make r stores device-visible before signaling
    __syncthreads();
    if (tid < Bsz) {
      atomicAdd(&out[tid * Ssz + t], lds_out[tid] * invH);
      lds_out[tid] = 0.f;
    }
    __syncthreads();

    // ---- device-wide step barrier (32 resident WGs)
    if (t + 1 < Ssz) {
      if (tid == 0) {
        __hip_atomic_fetch_add(gcount, 1u, __ATOMIC_RELEASE, __HIP_MEMORY_SCOPE_AGENT);
        const unsigned target = (unsigned)NWG * (unsigned)(t + 1);
        while (__hip_atomic_load(gcount, __ATOMIC_ACQUIRE, __HIP_MEMORY_SCOPE_AGENT) < target)
          __builtin_amdgcn_s_sleep(2);
      }
      __syncthreads();
    }
  }
}

extern "C" void kernel_launch(void* const* d_in, const int* in_sizes, int n_in,
                              void* d_out, int out_size, void* d_ws, size_t ws_size,
                              hipStream_t stream) {
  const float* x    = (const float*)d_in[0];   // [B,S,1]
  const float* nois = (const float*)d_in[1];   // [B,S,H]
  const float* wi   = (const float*)d_in[2];   // [H,1]
  const float* wo   = (const float*)d_in[3];   // [H,1]
  const float* A    = (const float*)d_in[4];   // [H,1]
  const float* rn   = (const float*)d_in[5];   // [H,H]
  const float* ovwi = (const float*)d_in[6];
  const float* ovwo = (const float*)d_in[7];
  const float* ovmn = (const float*)d_in[8];
  const float* h0   = (const float*)d_in[9];   // [H]
  float* out = (float*)d_out;                  // [B,S,1]

  // workspace layout
  char* ws = (char*)d_ws;
  float*    n_rec  = (float*)(ws);                        // 2048 f32
  float*    m_rec  = (float*)(ws + 8192);                 // 2048 f32
  __bf16*   rbuf   = (__bf16*)(ws + 16384);               // 2 x 64 x 2048 bf16
  unsigned* gcount = (unsigned*)(ws + 16384 + 2 * Bsz * Hsz * 2);

  rnn_setup<<<1, 256, 0, stream>>>(wi, wo, A, ovwi, ovwo, ovmn, h0,
                                   n_rec, m_rec, rbuf, out, gcount);
  rnn_persistent<<<NWG, 512, LDS_BYTES, stream>>>(x, nois, wi, wo, rn, h0,
                                                  n_rec, m_rec, rbuf, gcount, out);
}
